// ConvTran_618475290811
// MI455X (gfx1250) — compile-verified, matrix operand order checked by reading, not run on hardware
//
#include <hip/hip_runtime.h>
#include <math.h>
#include <stdint.h>

// ---------------------------------------------------------------------------
// ConvTran forward for MI455X (gfx1250, wave32, v_wmma_f32_16x16x32_f16).
// All WMMA operands are pre-converted f16 tensors padded so every A/B operand
// is two unconditional 16-byte global_load_b128 per lane (no guarded loads,
// no in-loop f32->f16 converts).  Attention uses the "S-transposed" online
// softmax so probabilities flow from WMMA C-layout straight into the next
// WMMA B-operand with zero cross-lane movement.
// ---------------------------------------------------------------------------

typedef __attribute__((ext_vector_type(16))) _Float16 v16h;
typedef __attribute__((ext_vector_type(8)))  _Float16 h8;
typedef __attribute__((ext_vector_type(8)))  float    v8f;

constexpr int Bn   = 8;
constexpr int Ln   = 1024;
constexpr int En   = 24;
constexpr int Hn   = 8;
constexpr int NCn  = 10;
constexpr int DFFn = 256;
constexpr int Kcv  = 8;
constexpr int HDn  = 3;

constexpr int NBLE = Bn * Ln * En;     // 196608
constexpr int NROW = Bn * Ln;          // 8192

// element j of a v16h A/B operand -> K index within the 32-wide K dimension:
// K = (j>>3)*16 + g*8 + (j&7)  ->  per lane: halves [g*8 .. g*8+7] and
// [16+g*8 .. 16+g*8+7] of the 32-padded K axis (two aligned b128 loads).
__device__ __forceinline__ v16h load_op32(const _Float16* base16B) {
    const h8* p = (const h8*)base16B;
    h8 lo = p[0];
    h8 hi = p[2];
    return __builtin_shufflevector(lo, hi, 0,1,2,3,4,5,6,7,8,9,10,11,12,13,14,15);
}

__device__ __forceinline__ v8f wmma_f16(v16h a, v16h b, v8f c) {
    return __builtin_amdgcn_wmma_f32_16x16x32_f16(false, a, false, b,
                                                  (short)0, c, false, false);
}

// ---------------------------------------------------------------------------
// Kernel 0a: zero-fill padded q/k/v f16 region (pads must be 0 every call)
// ---------------------------------------------------------------------------
__global__ void fill0_kernel(float4* __restrict__ p, int n16) {
    int i = blockIdx.x * blockDim.x + threadIdx.x;
    if (i < n16) p[i] = make_float4(0.f, 0.f, 0.f, 0.f);
}

// ---------------------------------------------------------------------------
// Kernel 0b: weight / bias-table preconversion to padded f16 layouts
//   wqT/wkT/wvT : [32 col][32 k]      (transposed, zero padded)
//   w1T         : [256 n][32 k]
//   w2T         : [32 col][256 k]
//   bt          : [8 h][2047 u], bt[h][u] = table[(2046-u)*H + h]
// ---------------------------------------------------------------------------
__global__ void prep_kernel(const float* __restrict__ wq, const float* __restrict__ wk,
                            const float* __restrict__ wv, const float* __restrict__ w1,
                            const float* __restrict__ w2, const float* __restrict__ tab,
                            _Float16* __restrict__ wqT, _Float16* __restrict__ wkT,
                            _Float16* __restrict__ wvT, _Float16* __restrict__ w1T,
                            _Float16* __restrict__ w2T, _Float16* __restrict__ bt) {
    int i = blockIdx.x * blockDim.x + threadIdx.x;
    if (i < 3 * 1024) {
        int wi = i >> 10, r = i & 1023;
        int col = r >> 5, kk = r & 31;
        const float* w = (wi == 0) ? wq : (wi == 1) ? wk : wv;
        _Float16*    o = (wi == 0) ? wqT : (wi == 1) ? wkT : wvT;
        o[col * 32 + kk] = (col < En && kk < En) ? (_Float16)w[kk * En + col]
                                                 : (_Float16)0.f;
        return;
    }
    i -= 3 * 1024;
    if (i < DFFn * 32) {
        int n = i >> 5, kk = i & 31;
        w1T[n * 32 + kk] = (kk < En) ? (_Float16)w1[kk * DFFn + n] : (_Float16)0.f;
        return;
    }
    i -= DFFn * 32;
    if (i < 32 * DFFn) {
        int col = i >> 8, kk = i & 255;
        w2T[col * 256 + kk] = (col < En) ? (_Float16)w2[kk * En + col] : (_Float16)0.f;
        return;
    }
    i -= 32 * DFFn;
    if (i < Hn * 2047) {
        int h = i / 2047, u = i % 2047;
        bt[h * 2047 + u] = (_Float16)tab[(2046 - u) * Hn + h];
        return;
    }
}

// ---------------------------------------------------------------------------
// Kernel 1: conv embed + BN(eval) + exact GELU + tAPE.
// Writes xsrc (f32, residual) and xposh (f16, row stride 32, zero padded).
// ---------------------------------------------------------------------------
__global__ void embed_kernel(const float* __restrict__ x,
                             const float* __restrict__ cw,
                             const float* __restrict__ cb,
                             const float* __restrict__ bg,
                             const float* __restrict__ bb,
                             const float* __restrict__ bm,
                             const float* __restrict__ bv,
                             float* __restrict__ xsrc,
                             _Float16* __restrict__ xposh) {
    int idx = blockIdx.x * blockDim.x + threadIdx.x;
    if (idx >= NBLE) return;
    int e   = idx % En;
    int row = idx / En;           // b*L + l
    int l   = row & (Ln - 1);
    int b   = row >> 10;

    float s = 0.f;
#pragma unroll
    for (int kk = 0; kk < Kcv; ++kk) {
        int t = l + kk - 3;
        if (t >= 0 && t < Ln) s += x[b * Ln + t] * cw[e * Kcv + kk];
    }
    s += cb[e];
    s = (s - bm[e]) * rsqrtf(bv[e] + 1e-5f) * bg[e] + bb[e];
    s = 0.5f * s * (1.f + erff(s * 0.70710678118654752f));
    xsrc[idx] = s;

    int   i2  = (e >> 1) * 2;
    float div = __expf((float)i2 * (-logf(10000.f) / (float)En));
    float ang = (float)l * div * ((float)En / (float)Ln);
    float pe  = (e & 1) ? cosf(ang) : sinf(ang);
    xposh[row * 32 + e] = (_Float16)(s + pe);
    if (e < 8) xposh[row * 32 + En + e] = (_Float16)0.f;   // zero the K pad
}

// ---------------------------------------------------------------------------
// Kernel 2: QKV projections (WMMA).  One wave per 16-row tile, 6 WMMAs.
// Outputs padded per-head f16: qh/kh (b,h,pos,32) [scale folded into qh],
// vt (b,h,16,L) transposed for the V^T A-operand.
// ---------------------------------------------------------------------------
__global__ void qkv_kernel(const _Float16* __restrict__ xposh,
                           const _Float16* __restrict__ wqT,
                           const _Float16* __restrict__ wkT,
                           const _Float16* __restrict__ wvT,
                           _Float16* __restrict__ qh,
                           _Float16* __restrict__ kh,
                           _Float16* __restrict__ vt) {
    int wave = (blockIdx.x * blockDim.x + threadIdx.x) >> 5;
    int lane = threadIdx.x & 31;
    if (wave >= NROW / 16) return;
    int m0 = wave * 16;
    int g  = lane >> 4;
    int mn = lane & 15;
    const float scale = rsqrtf((float)En);

    v16h a = load_op32(xposh + (m0 + mn) * 32 + g * 8);

    const _Float16* Ws[3] = {wqT, wkT, wvT};
#pragma unroll
    for (int wi = 0; wi < 3; ++wi) {
#pragma unroll
        for (int nt = 0; nt < 2; ++nt) {
            int col = nt * 16 + mn;
            v16h bmat = load_op32(Ws[wi] + col * 32 + g * 8);   // padded: no guard
            v8f c = {};
            c = wmma_f16(a, bmat, c);
            if (col < En) {
                int h = col / HDn, d = col % HDn;
#pragma unroll
                for (int r = 0; r < 8; ++r) {
                    int row = m0 + r + (g << 3);
                    int b   = row >> 10;
                    int pos = row & (Ln - 1);
                    int hb  = (b * Hn + h);
                    if (wi == 0)
                        qh[(hb * Ln + pos) * 32 + d] = (_Float16)(c[r] * scale);
                    else if (wi == 1)
                        kh[(hb * Ln + pos) * 32 + d] = (_Float16)c[r];
                    else
                        vt[(hb * 16 + d) * Ln + pos] = (_Float16)c[r];
                }
            }
        }
    }
}

// ---------------------------------------------------------------------------
// Kernel 3: fused attention (online softmax, post-softmax eRPE bias).
// One wave per (b,h,16-q tile); per 32-key chunk: 2 score WMMAs + PV WMMA +
// bias WMMA.  All operands are unconditional b128 loads except the Toeplitz
// bias gather (contiguous d16 loads).
// ---------------------------------------------------------------------------
__global__ void attn_kernel(const _Float16* __restrict__ qh,
                            const _Float16* __restrict__ kh,
                            const _Float16* __restrict__ vt,
                            const _Float16* __restrict__ bt,
                            float* __restrict__ ao) {
    int wave = (blockIdx.x * blockDim.x + threadIdx.x) >> 5;
    int lane = threadIdx.x & 31;
    if (wave >= Bn * Hn * (Ln / 16)) return;

    int qt = wave & 63;
    int h  = (wave >> 6) & (Hn - 1);
    int b  = wave >> 9;
    int q0 = qt * 16;
    int g  = lane >> 4;
    int ln = lane & 15;
    int hb = b * Hn + h;

    // Q^T B-operand (scale already folded in)
    v16h bq = load_op32(qh + (hb * Ln + q0 + ln) * 32 + g * 8);

    const _Float16* khb = kh + hb * Ln * 32;
    const _Float16* vtb = vt + hb * 16 * Ln;
    const _Float16* bth = bt + h * 2047;

    float mrow = -1e30f, lrow = 0.f;
    v8f acc  = {};
    v8f accB = {};

    for (int ch = 0; ch < Ln / 32; ++ch) {
        v8f st[2];
#pragma unroll
        for (int t = 0; t < 2; ++t) {
            v16h ak = load_op32(khb + (ch * 32 + t * 16 + ln) * 32 + g * 8);
            v8f z = {};
            st[t] = wmma_f16(ak, bq, z);
        }

        float cm = -1e30f;
#pragma unroll
        for (int r = 0; r < 8; ++r) cm = fmaxf(cm, fmaxf(st[0][r], st[1][r]));
        cm = fmaxf(cm, __shfl_xor(cm, 16, 32));
        float mnew  = fmaxf(mrow, cm);
        float alpha = __expf(mrow - mnew);

        float p0[8], p1[8];
        float rs = 0.f;
#pragma unroll
        for (int r = 0; r < 8; ++r) {
            p0[r] = __expf(st[0][r] - mnew);
            p1[r] = __expf(st[1][r] - mnew);
            rs += p0[r] + p1[r];
        }
        rs += __shfl_xor(rs, 16, 32);
        lrow = lrow * alpha + rs;
        mrow = mnew;
#pragma unroll
        for (int r = 0; r < 8; ++r) acc[r] *= alpha;

        // P^T: C-layout rows map 1:1 onto B-operand element slots
        v16h bp;
#pragma unroll
        for (int j = 0; j < 8; ++j) {
            bp[j]     = (_Float16)p0[j];
            bp[j + 8] = (_Float16)p1[j];
        }

        // V^T A-operand: two b128 loads from the transposed layout
        v16h av = load_op32(vtb + ln * Ln + ch * 32 + g * 8);
        acc = wmma_f16(av, bp, acc);

        // eRPE bias^T: contiguous (per-lane unaligned) gather from reversed table
        int u0 = ch * 32 + g * 8 - q0 - ln + 1023;
        v16h bbv;
#pragma unroll
        for (int j = 0; j < 8; ++j) {
            bbv[j]     = bth[u0 + j];
            bbv[j + 8] = bth[u0 + 16 + j];
        }
        accB = wmma_f16(av, bbv, accB);
    }

    float linv = 1.f / lrow;
    if (g == 0) {
#pragma unroll
        for (int r = 0; r < HDn; ++r)
            ao[(b * Ln + q0 + ln) * En + h * HDn + r] = acc[r] * linv + accB[r];
    }
}

// ---------------------------------------------------------------------------
// Kernel 4: to_out LN, residual, LN1.  Writes attf (f32 residual copy) and
// atth (padded f16 for the FFN1 A-operand).
// ---------------------------------------------------------------------------
__device__ __forceinline__ void ln_row(const float* in, float* out,
                                       const float* gam, const float* bet) {
    float mu = 0.f;
#pragma unroll
    for (int i = 0; i < En; ++i) mu += in[i];
    mu /= (float)En;
    float var = 0.f;
#pragma unroll
    for (int i = 0; i < En; ++i) { float d = in[i] - mu; var += d * d; }
    var /= (float)En;
    float inv = rsqrtf(var + 1e-5f);
#pragma unroll
    for (int i = 0; i < En; ++i) out[i] = (in[i] - mu) * inv * gam[i] + bet[i];
}

__global__ void postattn_kernel(const float* __restrict__ xsrc,
                                const float* __restrict__ ao,
                                const float* __restrict__ alg,
                                const float* __restrict__ alb,
                                const float* __restrict__ l1g,
                                const float* __restrict__ l1b,
                                float* __restrict__ attf,
                                _Float16* __restrict__ atth) {
    int row = blockIdx.x * blockDim.x + threadIdx.x;
    if (row >= NROW) return;
    float r0[En], r1[En];
#pragma unroll
    for (int i = 0; i < En; ++i) r0[i] = ao[row * En + i];
    ln_row(r0, r1, alg, alb);
#pragma unroll
    for (int i = 0; i < En; ++i) r1[i] += xsrc[row * En + i];
    ln_row(r1, r0, l1g, l1b);
#pragma unroll
    for (int i = 0; i < En; ++i) {
        attf[row * En + i]  = r0[i];
        atth[row * 32 + i] = (_Float16)r0[i];
    }
#pragma unroll
    for (int i = En; i < 32; ++i) atth[row * 32 + i] = (_Float16)0.f;
}

// ---------------------------------------------------------------------------
// Kernel 5: FFN1 (M=8192,N=256,K=24->32).  1 WMMA/wave; hid stored f16.
// ---------------------------------------------------------------------------
__global__ void ffn1_kernel(const _Float16* __restrict__ atth,
                            const _Float16* __restrict__ w1T,
                            const float* __restrict__ b1,
                            _Float16* __restrict__ hid) {
    int wave = (blockIdx.x * blockDim.x + threadIdx.x) >> 5;
    int lane = threadIdx.x & 31;
    if (wave >= (NROW / 16) * (DFFn / 16)) return;
    int mt = wave >> 4, nt = wave & 15;
    int m0 = mt * 16, n0 = nt * 16;
    int g = lane >> 4, mn = lane & 15;

    v16h a    = load_op32(atth + (m0 + mn) * 32 + g * 8);
    v16h bmat = load_op32(w1T + (n0 + mn) * 32 + g * 8);
    v8f c = {};
    c = wmma_f16(a, bmat, c);
    int   col = n0 + mn;
    float bia = b1[col];
#pragma unroll
    for (int r = 0; r < 8; ++r) {
        int row = m0 + r + (g << 3);
        hid[row * DFFn + col] = (_Float16)fmaxf(c[r] + bia, 0.f);
    }
}

// ---------------------------------------------------------------------------
// Kernel 6: FFN2 (M=8192,N=24->2 tiles,K=256) + bias + residual.
// ---------------------------------------------------------------------------
__global__ void ffn2_kernel(const _Float16* __restrict__ hid,
                            const _Float16* __restrict__ w2T,
                            const float* __restrict__ b2,
                            const float* __restrict__ attf,
                            float* __restrict__ pre) {
    int wave = (blockIdx.x * blockDim.x + threadIdx.x) >> 5;
    int lane = threadIdx.x & 31;
    if (wave >= (NROW / 16) * 2) return;
    int mt = wave >> 1, nt = wave & 1;
    int m0 = mt * 16, n0 = nt * 16;
    int g = lane >> 4, mn = lane & 15;
    int col = n0 + mn;

    v8f c = {};
    for (int kc = 0; kc < DFFn / 32; ++kc) {
        int k0 = kc * 32;
        v16h a    = load_op32(hid + (m0 + mn) * DFFn + k0 + g * 8);
        v16h bmat = load_op32(w2T + col * 256 + k0 + g * 8);    // padded cols = 0
        c = wmma_f16(a, bmat, c);
    }
    if (col < En) {
        float bia = b2[col];
#pragma unroll
        for (int r = 0; r < 8; ++r) {
            int row = m0 + r + (g << 3);
            pre[row * En + col] = c[r] + bia + attf[row * En + col];
        }
    }
}

// ---------------------------------------------------------------------------
// Kernel 7: LN2 + mean-pool over L + classifier.  One block per batch.
// ---------------------------------------------------------------------------
__global__ void head_kernel(const float* __restrict__ pre,
                            const float* __restrict__ l2g,
                            const float* __restrict__ l2b,
                            const float* __restrict__ ow,
                            const float* __restrict__ ob,
                            float* __restrict__ out) {
    __shared__ float sacc[En];
    int b = blockIdx.x;
    int tid = threadIdx.x;
    if (tid < En) sacc[tid] = 0.f;
    __syncthreads();

    float local[En];
#pragma unroll
    for (int i = 0; i < En; ++i) local[i] = 0.f;

    for (int l = tid; l < Ln; l += blockDim.x) {
        float row[En], o[En];
#pragma unroll
        for (int i = 0; i < En; ++i) row[i] = pre[(b * Ln + l) * En + i];
        ln_row(row, o, l2g, l2b);
#pragma unroll
        for (int i = 0; i < En; ++i) local[i] += o[i];
    }
#pragma unroll
    for (int i = 0; i < En; ++i) atomicAdd(&sacc[i], local[i]);
    __syncthreads();

    if (tid < NCn) {
        float s = ob[tid];
#pragma unroll
        for (int i = 0; i < En; ++i)
            s += (sacc[i] / (float)Ln) * ow[i * NCn + tid];
        out[b * NCn + tid] = s;
    }
}

// ---------------------------------------------------------------------------
// Launch
// ---------------------------------------------------------------------------
extern "C" void kernel_launch(void* const* d_in, const int* in_sizes, int n_in,
                              void* d_out, int out_size, void* d_ws, size_t ws_size,
                              hipStream_t stream) {
    (void)in_sizes; (void)n_in; (void)out_size; (void)ws_size;

    const float* x      = (const float*)d_in[0];
    const float* conv_w = (const float*)d_in[1];
    const float* conv_b = (const float*)d_in[2];
    const float* bn_g   = (const float*)d_in[3];
    const float* bn_b   = (const float*)d_in[4];
    const float* bn_m   = (const float*)d_in[5];
    const float* bn_v   = (const float*)d_in[6];
    const float* wq     = (const float*)d_in[7];
    const float* wk     = (const float*)d_in[8];
    const float* wv     = (const float*)d_in[9];
    const float* rtab   = (const float*)d_in[10];
    const float* alg    = (const float*)d_in[11];
    const float* alb    = (const float*)d_in[12];
    const float* l1g    = (const float*)d_in[13];
    const float* l1b    = (const float*)d_in[14];
    const float* l2g    = (const float*)d_in[15];
    const float* l2b    = (const float*)d_in[16];
    const float* ff_w1  = (const float*)d_in[17];
    const float* ff_b1  = (const float*)d_in[18];
    const float* ff_w2  = (const float*)d_in[19];
    const float* ff_b2  = (const float*)d_in[20];
    const float* out_w  = (const float*)d_in[21];
    const float* out_b  = (const float*)d_in[22];
    float* out = (float*)d_out;

    // ---- workspace carve (f32 region, then 16B-aligned f16 region) --------
    float* fw   = (float*)d_ws;
    float* xsrc = fw;                   // NBLE
    float* ao   = xsrc + NBLE;          // NBLE
    float* attf = ao + NBLE;            // NBLE
    float* pre  = attf + NBLE;          // NBLE

    _Float16* hw   = (_Float16*)(pre + NBLE);          // 16B aligned by construction
    _Float16* xposh = hw;                               // NROW*32
    _Float16* qh    = xposh + NROW * 32;                // B*H*L*32
    _Float16* kh    = qh + Bn * Hn * Ln * 32;           // B*H*L*32
    _Float16* vt    = kh + Bn * Hn * Ln * 32;           // B*H*16*L
    _Float16* atth  = vt + Bn * Hn * 16 * Ln;           // NROW*32
    _Float16* hid   = atth + NROW * 32;                 // NROW*256
    _Float16* wqT   = hid + NROW * DFFn;                // 1024
    _Float16* wkT   = wqT + 1024;
    _Float16* wvT   = wkT + 1024;
    _Float16* w1T   = wvT + 1024;                       // 256*32
    _Float16* w2T   = w1T + DFFn * 32;                  // 32*256
    _Float16* bt    = w2T + 32 * DFFn;                  // 8*2047

    // zero the padded q/k/v region (qh,kh,vt contiguous)
    const int fillHalves = Bn * Hn * Ln * 32 * 2 + Bn * Hn * 16 * Ln;
    const int fill16     = fillHalves * 2 / 16;
    fill0_kernel<<<(fill16 + 255) / 256, 256, 0, stream>>>((float4*)qh, fill16);

    const int prepN = 3 * 1024 + DFFn * 32 + 32 * DFFn + Hn * 2047;
    prep_kernel<<<(prepN + 255) / 256, 256, 0, stream>>>(
        wq, wk, wv, ff_w1, ff_w2, rtab, wqT, wkT, wvT, w1T, w2T, bt);

    embed_kernel<<<(NBLE + 255) / 256, 256, 0, stream>>>(
        x, conv_w, conv_b, bn_g, bn_b, bn_m, bn_v, xsrc, xposh);

    qkv_kernel<<<(NROW / 16) * 32 / 256, 256, 0, stream>>>(
        xposh, wqT, wkT, wvT, qh, kh, vt);

    attn_kernel<<<(Bn * Hn * (Ln / 16)) * 32 / 256, 256, 0, stream>>>(
        qh, kh, vt, bt, ao);

    postattn_kernel<<<(NROW + 255) / 256, 256, 0, stream>>>(
        xsrc, ao, alg, alb, l1g, l1b, attf, atth);

    ffn1_kernel<<<((NROW / 16) * (DFFn / 16)) * 32 / 256, 256, 0, stream>>>(
        atth, w1T, ff_b1, hid);

    ffn2_kernel<<<((NROW / 16) * 2) * 32 / 256, 256, 0, stream>>>(
        hid, w2T, ff_b2, attf, pre);

    head_kernel<<<Bn, 128, 0, stream>>>(pre, l2g, l2b, out_w, out_b, out);
}
